// Insertnerf_51565377355936
// MI455X (gfx1250) — matrix operand order repaired and checked by reading, not compiled, not run on hardware
//
#include <hip/hip_runtime.h>

typedef _Float16 half_t;
typedef half_t v16h __attribute__((ext_vector_type(16)));
typedef half_t v8h  __attribute__((ext_vector_type(8)));
typedef float  v8f  __attribute__((ext_vector_type(8)));

#define DEV __device__ __forceinline__

DEV v8f wmma_f16(v16h a, v16h b, v8f c) {
  // D = A(16x32 f16) * B(32x16 f16) + C(16x16 f32)
  return __builtin_amdgcn_wmma_f32_16x16x32_f16(false, a, false, b, (short)0, c,
                                                false, false);
}

DEV float eluf(float x) { return x > 0.f ? x : (__expf(x) - 1.f); }
DEV float sigf(float x) { return 1.f / (1.f + __expf(-x)); }
DEV void wave_sync() { __builtin_amdgcn_wave_barrier(); }

// ---- WMMA operand loaders (CDNA5 wave32 layouts) -------------------------
// A 16x32 f16: lane L<16 holds row M=L with K = {0..7, 16..23}; lane L+16
// holds row M=L with K = {8..15, 24..31}. Two contiguous 8-half runs ->
// compiler emits 2x ds_load_b128.
DEV v16h load_a(const half_t* act, int Wa, int lane, int kt) {
  const int m = lane & 15;
  const int kbase = kt * 32 + ((lane >> 4) ? 8 : 0);
  v16h a;
#pragma unroll
  for (int i = 0; i < 16; i++) {
    int k = kbase + ((i < 8) ? i : (8 + i));
    a[i] = act[m * Wa + k];
  }
  return a;
}

// B operand: weights pre-swizzled in LDS into fragment order
// [tile = kt*NT+nt][lane][16 halves]: one aligned 32-byte LDS read per lane.
template <int KT, int NT>
DEV void gemm(const half_t* A, int Wa, const half_t* Bfrag, int lane, v8f* c) {
#pragma unroll
  for (int nt = 0; nt < NT; nt++) {
    v8f z = {};
    c[nt] = z;
  }
  const v16h* bf = (const v16h*)Bfrag;
#pragma unroll
  for (int kt = 0; kt < KT; kt++) {
    v16h a = load_a(A, Wa, lane, kt);
#pragma unroll
    for (int nt = 0; nt < NT; nt++) {
      v16h b = bf[(kt * NT + nt) * 32 + lane];
      c[nt] = wmma_f16(a, b, c[nt]);
    }
  }
}

enum { ACT_NONE = 0, ACT_RELU = 1, ACT_ELU = 2 };

template <int NT, int MODE>
DEV void bias_act(v8f* c, const float* bias, int lane) {
  const int n = lane & 15;
#pragma unroll
  for (int nt = 0; nt < NT; nt++) {
    float b = bias[nt * 16 + n];
#pragma unroll
    for (int k = 0; k < 8; k++) {
      float v = c[nt][k] + b;
      if (MODE == ACT_RELU) v = fmaxf(v, 0.f);
      if (MODE == ACT_ELU) v = eluf(v);
      c[nt][k] = v;
    }
  }
}

// D layout: vgpr k holds row M = k + 8*(lane>=16); N = nt*16 + (lane&15)
template <int NT>
DEV void store_tiles(half_t* dst, int Wd, const v8f* c, int lane) {
  const int n = lane & 15, h = lane >> 4;
#pragma unroll
  for (int nt = 0; nt < NT; nt++)
#pragma unroll
    for (int k = 0; k < 8; k++)
      dst[(k + 8 * h) * Wd + nt * 16 + n] = (half_t)c[nt][k];
}

// uniform, fully-unrolled zero fill; COUNT (halves) % 256 == 0
template <int COUNT>
DEV void zero_buf(half_t* p, int lane) {
  static_assert((COUNT % 256) == 0, "count must be multiple of 256 halves");
  uint4* q = (uint4*)p;
  uint4 z;
  z.x = z.y = z.z = z.w = 0u;
#pragma unroll
  for (int t = 0; t < COUNT / 256; t++) q[t * 32 + lane] = z;
}

// ---- LDS layout -----------------------------------------------------------
struct WaveScratch {
  half_t bufA[16 * 224];  // A operand (max width 224 for base layer)
  half_t bufB[16 * 64];   // intermediate A operand
  half_t keep[16 * 68];   // cols 0..34 rgb_feat_, 35..66 neuray_feat
  float rsc_mask[16], rsc_w[16], rsc_w0[16], rsc_vis[16], rsc_vis2[16];
  float fmax[2 * 64];     // rgb_feat_max per sample of the pair
};

struct __align__(32) SharedMem {
  // weights in B-fragment order: [KT*NT][32 lanes][16 halves]
  half_t w_rd1[1 * 1 * 512];  half_t w_rd2[1 * 3 * 512];
  half_t w_rf1[2 * 4 * 512];  half_t w_rf2[2 * 4 * 512];
  half_t w_nr1[1 * 1 * 512];  half_t w_nr2[1 * 1 * 512];
  half_t w_b1[7 * 4 * 512];   half_t w_b2[2 * 2 * 512];
  half_t w_v1[1 * 2 * 512];   half_t w_v2[1 * 3 * 512];
  half_t w_q1[1 * 2 * 512];   half_t w_q2[1 * 1 * 512];
  float b_rd1[16], b_rd2[48], b_rf1[64], b_rf2[64];
  float b_nr1[16], b_nr2[16], b_b1[64], b_b2[32];
  float b_v1[32], b_v2[48], b_q1[32], b_q2[16];
  WaveScratch ws[4];
};

// Stage weight [K x N] (f32, row-major) into LDS as zero-padded f16
// B-fragments. Each thread produces one 8-half run (16 B store).
DEV void stage_w_frag(half_t* dst, const float* src, int K, int N, int NT,
                      int totalHalves, int tid) {
  for (int base = tid * 8; base < totalHalves; base += 128 * 8) {
    const int tile = base >> 9;
    const int within = base & 511;
    const int ln = within >> 4;
    const int i0 = within & 15;  // 0 or 8
    const int kt = tile / NT, nt = tile - kt * NT;
    const int col = nt * 16 + (ln & 15);
    const int kbase = kt * 32 + ((ln >= 16) ? 8 : 0) + (i0 ? 16 : 0);
    v8h v;
#pragma unroll
    for (int j = 0; j < 8; j++) {
      int k = kbase + j;
      v[j] = (half_t)((k < K && col < N) ? src[k * N + col] : 0.f);
    }
    *(v8h*)(dst + base) = v;
  }
}
DEV void stage_b(float* dst, const float* src, int N, int Np, int tid) {
  for (int i = tid; i < Np; i += 128) dst[i] = (i < N) ? src[i] : 0.f;
}

__global__ __launch_bounds__(128, 1) void nerf_agg_kernel(
    const float* __restrict__ rgb, const float* __restrict__ neur,
    const float* __restrict__ rdif, const float* __restrict__ maskp,
    const float* rd_w1, const float* rd_b1, const float* rd_w2,
    const float* rd_b2, const float* rf_w1, const float* rf_b1,
    const float* rf_w2, const float* rf_b2, const float* nr_w1,
    const float* nr_b1, const float* nr_w2, const float* nr_b2,
    const float* ba_w1, const float* ba_b1, const float* ba_w2,
    const float* ba_b2, const float* vi_w1, const float* vi_b1,
    const float* vi_w2, const float* vi_b2, const float* q_w1,
    const float* q_b1, const float* q_w2, const float* q_b2,
    float* __restrict__ out, int nSamples) {
  __shared__ SharedMem sh;
  const int tid = threadIdx.x;

  // ---- stage all weights as swizzled f16 B-fragments + biases (once) ----
  stage_w_frag(sh.w_rd1, rd_w1, 4, 16, 1, 1 * 1 * 512, tid);
  stage_w_frag(sh.w_rd2, rd_w2, 16, 35, 3, 1 * 3 * 512, tid);
  stage_w_frag(sh.w_rf1, rf_w1, 35, 64, 4, 2 * 4 * 512, tid);
  stage_w_frag(sh.w_rf2, rf_w2, 64, 64, 4, 2 * 4 * 512, tid);
  stage_w_frag(sh.w_nr1, nr_w1, 32, 8, 1, 1 * 1 * 512, tid);
  stage_w_frag(sh.w_nr2, nr_w2, 8, 1, 1, 1 * 1 * 512, tid);
  stage_w_frag(sh.w_b1, ba_w1, 207, 64, 4, 7 * 4 * 512, tid);
  stage_w_frag(sh.w_b2, ba_w2, 64, 32, 2, 2 * 2 * 512, tid);
  stage_w_frag(sh.w_v1, vi_w1, 32, 32, 2, 1 * 2 * 512, tid);
  stage_w_frag(sh.w_v2, vi_w2, 32, 33, 3, 1 * 3 * 512, tid);
  stage_w_frag(sh.w_q1, q_w1, 32, 32, 2, 1 * 2 * 512, tid);
  stage_w_frag(sh.w_q2, q_w2, 32, 1, 1, 1 * 1 * 512, tid);
  stage_b(sh.b_rd1, rd_b1, 16, 16, tid);
  stage_b(sh.b_rd2, rd_b2, 35, 48, tid);
  stage_b(sh.b_rf1, rf_b1, 64, 64, tid);
  stage_b(sh.b_rf2, rf_b2, 64, 64, tid);
  stage_b(sh.b_nr1, nr_b1, 8, 16, tid);
  stage_b(sh.b_nr2, nr_b2, 1, 16, tid);
  stage_b(sh.b_b1, ba_b1, 64, 64, tid);
  stage_b(sh.b_b2, ba_b2, 32, 32, tid);
  stage_b(sh.b_v1, vi_b1, 32, 32, tid);
  stage_b(sh.b_v2, vi_b2, 33, 48, tid);
  stage_b(sh.b_q1, q_b1, 32, 32, tid);
  stage_b(sh.b_q2, q_b2, 1, 16, tid);
  __syncthreads();

  const int lane = tid & 31;
  const int wv = tid >> 5;
  WaveScratch& S = sh.ws[wv];
  const int n = lane & 15;  // feature-within-tile
  const int g = lane >> 4;  // which sample of the pair this lane reduces

  const int pairsTotal = (nSamples + 1) >> 1;
  const int pairStride = gridDim.x * 4;

  // persistent grid-stride loop over sample pairs: weights staged once
  for (int pair = blockIdx.x * 4 + wv; pair < pairsTotal; pair += pairStride) {
    const int gs0 = pair * 2;  // first sample of pair
    const int p0 = gs0 * 8;    // first (sample,view) row

    // ---- mask -> weight (per row) ----
    if (lane < 16) S.rsc_mask[lane] = maskp[p0 + lane];
    wave_sync();
    if (lane < 16) {
      int gg = lane >> 3;
      float s = 0.f;
#pragma unroll
      for (int j = 0; j < 8; j++) s += S.rsc_mask[gg * 8 + j];
      S.rsc_w[lane] = S.rsc_mask[lane] / (s + 1e-8f);
    }
    wave_sync();

    // ---- neuray_feat -> keep cols 35..66 (512 elems, uniform) ----
#pragma unroll
    for (int t = 0; t < 16; t++) {
      int i = t * 32 + lane;
      int r = i >> 5, j = i & 31;
      S.keep[r * 68 + 35 + j] = (half_t)neur[(p0 + r) * 32 + j];
    }
    wave_sync();

    // ---- neuray MLP -> weight0 ----
    {
      v8f c[1];
      gemm<1, 1>(S.keep + 35, 68, sh.w_nr1, lane, c);
      bias_act<1, ACT_ELU>(c, sh.b_nr1, lane);
      zero_buf<16 * 32>(S.bufB, lane);
      store_tiles<1>(S.bufB, 32, c, lane);
      gemm<1, 1>(S.bufB, 32, sh.w_nr2, lane, c);
      if (n == 0) {
#pragma unroll
        for (int k = 0; k < 8; k++)
          S.rsc_w0[g * 8 + k] = sigf(c[0][k] + sh.b_nr2[0]);
      }
      wave_sync();
    }

    // ---- ray_diff MLP -> direction_feat (kept in regs, D layout) ----
    v8f cdir[3];
    {
      zero_buf<16 * 32>(S.bufB, lane);
#pragma unroll
      for (int t = 0; t < 2; t++) {
        int i = t * 32 + lane;
        int r = i >> 2, j = i & 3;
        S.bufB[r * 32 + j] = (half_t)rdif[(p0 + r) * 4 + j];
      }
      v8f c[1];
      gemm<1, 1>(S.bufB, 32, sh.w_rd1, lane, c);
      bias_act<1, ACT_ELU>(c, sh.b_rd1, lane);
      zero_buf<16 * 32>(S.bufB, lane);
      store_tiles<1>(S.bufB, 32, c, lane);
      gemm<1, 3>(S.bufB, 32, sh.w_rd2, lane, cdir);
      bias_act<3, ACT_ELU>(cdir, sh.b_rd2, lane);
    }

    // ---- rgb_feat -> bufA (padded W=64); rgb_feat_ + weighted mean/var ----
    zero_buf<16 * 64>(S.bufA, lane);
#pragma unroll
    for (int t = 0; t < 17; t++) {  // 17*32 = 544 of 560
      int i = t * 32 + lane;
      int r = i / 35, j = i - r * 35;
      S.bufA[r * 64 + j] = (half_t)rgb[(p0 + r) * 35 + j];
    }
    if (lane < 16) {  // tail 544..559
      int i = 544 + lane;
      int r = i / 35, j = i - r * 35;
      S.bufA[r * 64 + j] = (half_t)rgb[(p0 + r) * 35 + j];
    }
    wave_sync();

    float w0r[8], wr[8];
#pragma unroll
    for (int k = 0; k < 8; k++) {
      w0r[k] = S.rsc_w0[g * 8 + k];
      wr[k] = S.rsc_w[g * 8 + k];
    }
    float xv[3][8];
    float m0[3] = {0.f, 0.f, 0.f}, m1[3] = {0.f, 0.f, 0.f}, v0[3], v1[3];
#pragma unroll
    for (int nt = 0; nt < 3; nt++) {
      int f = nt * 16 + n;
#pragma unroll
      for (int k = 0; k < 8; k++) {
        float x = 0.f;
        if (f < 35) {
          x = (float)S.bufA[(k + 8 * g) * 64 + f] + cdir[nt][k];
          S.keep[(k + 8 * g) * 68 + f] = (half_t)x;  // persist rgb_feat_
        }
        xv[nt][k] = x;
        m0[nt] += w0r[k] * x;
        m1[nt] += wr[k] * x;
      }
    }
#pragma unroll
    for (int nt = 0; nt < 3; nt++) {
      float a0 = 0.f, a1 = 0.f;
#pragma unroll
      for (int k = 0; k < 8; k++) {
        float d0 = xv[nt][k] - m0[nt], d1 = xv[nt][k] - m1[nt];
        a0 += w0r[k] * d0 * d0;
        a1 += wr[k] * d1 * d1;
      }
      v0[nt] = a0;
      v1[nt] = a1;
    }
    wave_sync();

    // ---- rgb MLP (on raw rgb_feat) -> rgb_feat_max ----
    {
      v8f c[4];
      gemm<2, 4>(S.bufA, 64, sh.w_rf1, lane, c);
      bias_act<4, ACT_RELU>(c, sh.b_rf1, lane);
      zero_buf<16 * 64>(S.bufB, lane);
      store_tiles<4>(S.bufB, 64, c, lane);
      gemm<2, 4>(S.bufB, 64, sh.w_rf2, lane, c);
      bias_act<4, ACT_NONE>(c, sh.b_rf2, lane);
#pragma unroll
      for (int nt = 0; nt < 4; nt++) {
        float mx = c[nt][0];
#pragma unroll
        for (int k = 1; k < 8; k++) mx = fmaxf(mx, c[nt][k]);
        S.fmax[g * 64 + nt * 16 + n] = mx;
      }
      wave_sync();
    }

    // ---- build concat [globalfeat(140) | rgb_feat_(35) | neuray(32)] ----
    zero_buf<16 * 224>(S.bufA, lane);
#pragma unroll
    for (int nt = 0; nt < 3; nt++) {
      int f = nt * 16 + n;
      if (f < 35) {
#pragma unroll
        for (int r = 0; r < 8; r++) {
          half_t* row = &S.bufA[(g * 8 + r) * 224];
          row[f] = (half_t)m0[nt];
          row[35 + f] = (half_t)v0[nt];
          row[70 + f] = (half_t)m1[nt];
          row[105 + f] = (half_t)v1[nt];
        }
      }
    }
#pragma unroll
    for (int t = 0; t < 33; t++) {  // 33*32 = 1056 of 1072
      int i = t * 32 + lane;
      int r = i / 67, j = i - r * 67;
      S.bufA[r * 224 + 140 + j] = S.keep[r * 68 + j];
    }
    if (lane < 16) {  // tail 1056..1071
      int i = 1056 + lane;
      int r = i / 67, j = i - r * 67;
      S.bufA[r * 224 + 140 + j] = S.keep[r * 68 + j];
    }
    wave_sync();

    // ---- base MLP: 207(->224) -> 64 -> 32 ; x kept in regs ----
    v8f cx[2];
    {
      v8f c[4];
      gemm<7, 4>(S.bufA, 224, sh.w_b1, lane, c);
      bias_act<4, ACT_ELU>(c, sh.b_b1, lane);
      zero_buf<16 * 64>(S.bufB, lane);
      store_tiles<4>(S.bufB, 64, c, lane);
      gemm<2, 2>(S.bufB, 64, sh.w_b2, lane, cx);
      bias_act<2, ACT_ELU>(cx, sh.b_b2, lane);
    }

    // ---- vis branch: x*weight -> 32 -> 33 ; residual + vis ----
    {
#pragma unroll
      for (int nt = 0; nt < 2; nt++)
#pragma unroll
        for (int k = 0; k < 8; k++)
          S.bufB[(k + 8 * g) * 32 + nt * 16 + n] = (half_t)(cx[nt][k] * wr[k]);
      wave_sync();
      v8f c2[2];
      gemm<1, 2>(S.bufB, 32, sh.w_v1, lane, c2);
      bias_act<2, ACT_ELU>(c2, sh.b_v1, lane);
      store_tiles<2>(S.bufB, 32, c2, lane);
      wave_sync();
      v8f c3[3];
      gemm<1, 3>(S.bufB, 32, sh.w_v2, lane, c3);
      bias_act<3, ACT_ELU>(c3, sh.b_v2, lane);
#pragma unroll
      for (int nt = 0; nt < 2; nt++)
#pragma unroll
        for (int k = 0; k < 8; k++) cx[nt][k] += c3[nt][k];  // x += x_res
      if (n == 0) {  // col 32 lives in tile 2, n==0
#pragma unroll
        for (int k = 0; k < 8; k++)
          S.rsc_vis[g * 8 + k] = sigf(c3[2][k]) * S.rsc_mask[g * 8 + k];
      }
      wave_sync();
    }

    // ---- vis2 branch: x*vis -> 32 -> 1 -> sigmoid*mask ----
    {
      float visr[8];
#pragma unroll
      for (int k = 0; k < 8; k++) visr[k] = S.rsc_vis[g * 8 + k];
#pragma unroll
      for (int nt = 0; nt < 2; nt++)
#pragma unroll
        for (int k = 0; k < 8; k++)
          S.bufB[(k + 8 * g) * 32 + nt * 16 + n] =
              (half_t)(cx[nt][k] * visr[k]);
      wave_sync();
      v8f c2[2];
      gemm<1, 2>(S.bufB, 32, sh.w_q1, lane, c2);
      bias_act<2, ACT_ELU>(c2, sh.b_q1, lane);
      zero_buf<16 * 32>(S.bufB, lane);
      store_tiles<2>(S.bufB, 32, c2, lane);
      wave_sync();
      v8f c1[1];
      gemm<1, 1>(S.bufB, 32, sh.w_q2, lane, c1);
      if (n == 0) {
#pragma unroll
        for (int k = 0; k < 8; k++)
          S.rsc_vis2[g * 8 + k] =
              sigf(c1[0][k] + sh.b_q2[0]) * S.rsc_mask[g * 8 + k];
      }
      wave_sync();
    }

    // ---- weight2-normalized mean/var over views + rgb_feat_max -> out ----
    {
      float v2r[8], s = 0.f;
#pragma unroll
      for (int k = 0; k < 8; k++) {
        v2r[k] = S.rsc_vis2[g * 8 + k];
        s += v2r[k];
      }
      float inv = 1.f / (s + 1e-8f);
#pragma unroll
      for (int k = 0; k < 8; k++) v2r[k] *= inv;
      const int gs = gs0 + g;
#pragma unroll
      for (int nt = 0; nt < 2; nt++) {
        float m = 0.f;
#pragma unroll
        for (int k = 0; k < 8; k++) m += v2r[k] * cx[nt][k];
        float va = 0.f;
#pragma unroll
        for (int k = 0; k < 8; k++) {
          float d = cx[nt][k] - m;
          va += v2r[k] * d * d;
        }
        int f = nt * 16 + n;
        out[gs * 64 + f] = m + S.fmax[g * 64 + f];
        out[gs * 64 + 32 + f] = va + S.fmax[g * 64 + 32 + f];
      }
    }
  }  // pair loop
}

extern "C" void kernel_launch(void* const* d_in, const int* in_sizes, int n_in,
                              void* d_out, int out_size, void* d_ws,
                              size_t ws_size, hipStream_t stream) {
  (void)n_in; (void)d_ws; (void)ws_size; (void)out_size;
  const float* rgb = (const float*)d_in[0];
  const float* neur = (const float*)d_in[1];
  const float* rdif = (const float*)d_in[2];
  const float* mask = (const float*)d_in[3];
  // d_in[4] que_pts, d_in[5] que_dir are unused by the reference
  const float* rd_w1 = (const float*)d_in[6];
  const float* rd_b1 = (const float*)d_in[7];
  const float* rd_w2 = (const float*)d_in[8];
  const float* rd_b2 = (const float*)d_in[9];
  const float* rf_w1 = (const float*)d_in[10];
  const float* rf_b1 = (const float*)d_in[11];
  const float* rf_w2 = (const float*)d_in[12];
  const float* rf_b2 = (const float*)d_in[13];
  const float* nr_w1 = (const float*)d_in[14];
  const float* nr_b1 = (const float*)d_in[15];
  const float* nr_w2 = (const float*)d_in[16];
  const float* nr_b2 = (const float*)d_in[17];
  const float* ba_w1 = (const float*)d_in[18];
  const float* ba_b1 = (const float*)d_in[19];
  const float* ba_w2 = (const float*)d_in[20];
  const float* ba_b2 = (const float*)d_in[21];
  const float* vi_w1 = (const float*)d_in[22];
  const float* vi_b1 = (const float*)d_in[23];
  const float* vi_w2 = (const float*)d_in[24];
  const float* vi_b2 = (const float*)d_in[25];
  const float* q_w1 = (const float*)d_in[26];
  const float* q_b1 = (const float*)d_in[27];
  const float* q_w2 = (const float*)d_in[28];
  const float* q_b2 = (const float*)d_in[29];

  const int nSamples = in_sizes[0] / (8 * 35);  // R*S
  const int pairs = (nSamples + 1) / 2;
  int blocks = (pairs + 3) / 4;
  if (blocks > 2048) blocks = 2048;  // persistent waves, grid-stride on pairs

  nerf_agg_kernel<<<blocks, 128, 0, stream>>>(
      rgb, neur, rdif, mask, rd_w1, rd_b1, rd_w2, rd_b2, rf_w1, rf_b1, rf_w2,
      rf_b2, nr_w1, nr_b1, nr_w2, nr_b2, ba_w1, ba_b1, ba_w2, ba_b2, vi_w1,
      vi_b1, vi_w2, vi_b2, q_w1, q_b1, q_w2, q_b2, (float*)d_out, nSamples);
}
